// VectorQuantizer_90787018703681
// MI455X (gfx1250) — compile-verified, matrix operand order checked by reading, not run on hardware
//
#include <hip/hip_runtime.h>
#include <hip/hip_bf16.h>

// ---------------------------------------------------------------------------
// VQ-VAE vector quantizer for MI455X (gfx1250, wave32, WMMA bf16).
//   z:   [131072, 256] fp32      emb: [512, 256] fp32
//   out: z_q (33554432 f32) | vq_loss | commit_loss | idx (131072, as f32)
// Fused GEMM+argmin via V_WMMA_F32_16X16X32_BF16. Codebook staged into LDS
// pre-swizzled into B-fragment layout -> conflict-free ds_load_b128 in the
// hot loop. Gather + loss in exact fp32. Deterministic reductions.
// ---------------------------------------------------------------------------

#define E_DIM   256
#define N_CODES 512
#define N_ROWS  131072            // 32*16*16*16
#define BETA    0.25f
#define ZQ_ELEMS 33554432         // N_ROWS * E_DIM

#define CODES_PER_STAGE 128       // 64 KB of bf16 per stage
#define N_STAGES (N_CODES / CODES_PER_STAGE)
#define STAGE_U4 4096             // 64 KB / 16 B

typedef __attribute__((ext_vector_type(16))) __bf16 v16bf;
typedef __attribute__((ext_vector_type(8)))  __bf16 v8bf;
typedef __attribute__((ext_vector_type(8)))  float  v8f;

// workspace layout (bytes):
//   [0,      262144) : codebook as bf16 (512*256)
//   [262144, 264192) : ||e||^2 per code (512 f32)
//   [264192, 268288) : per-block loss partials (1024 f32)
#define WS_ENORM_OFF 262144
#define WS_PART_OFF  264192

// ---- prep: fp32 codebook -> bf16, and per-code squared norms --------------
__global__ void vq_prep_90787018703681(const float* __restrict__ emb,
                                       __bf16* __restrict__ cb,
                                       float* __restrict__ enorm) {
  __shared__ float red[256];
  const int c = blockIdx.x;      // one block per code
  const int t = threadIdx.x;     // one thread per element
  float v = emb[c * E_DIM + t];
  cb[c * E_DIM + t] = (__bf16)v;
  red[t] = v * v;
  __syncthreads();
  for (int s = 128; s > 0; s >>= 1) {
    if (t < s) red[t] += red[t + s];
    __syncthreads();
  }
  if (t == 0) enorm[c] = red[0];
}

// ---- main: fused GEMM + argmin + gather + loss partial --------------------
__global__ void __launch_bounds__(256)
vq_main_90787018703681(const float* __restrict__ z,
                       const float* __restrict__ emb,
                       const __bf16* __restrict__ cb,
                       const float* __restrict__ enorm,
                       float* __restrict__ zq,
                       float* __restrict__ idx_out,
                       float* __restrict__ partials) {
  // 64 KB LDS: one stage of B fragments, layout
  //   frag[(ctl*8 + kc)*2 + chunk] = 32 lanes x 16 B   (lane stride 16 B:
  //   lanes 0..15 cover all 64 banks -> conflict-free ds_load_b128)
  __shared__ __align__(16) uint4 lds_b[STAGE_U4];
  __shared__ float wsum[8];

  const int lane = threadIdx.x & 31;
  const int wave = threadIdx.x >> 5;
  const int hi   = lane >> 4;          // lane half (K interleave select)
  const int lo   = lane & 15;          // row (A) / column (B) index
  const int rowBase = blockIdx.x * 128 + wave * 16;   // 16 rows per wave

  // --- A fragments: this wave's 16 z-rows, bf16, 8 chunks of K=32.
  // ISA 7.12.2 16-bit A 16x32: lane<16 row M=lane, elems0-7 <- K 0..7,
  // elems8-15 <- K 16..23; lane>=16 shifts K by +8 in each group.
  v16bf A[8];
  {
    const float* zr = z + (size_t)(rowBase + lo) * E_DIM;
#pragma unroll
    for (int kc = 0; kc < 8; ++kc) {
      const int k0 = kc * 32 + hi * 8;
#pragma unroll
      for (int i = 0; i < 8; ++i) {
        A[kc][i]     = (__bf16)zr[k0 + i];
        A[kc][8 + i] = (__bf16)zr[k0 + 16 + i];
      }
    }
  }

  float minval[8];
  int   minidx[8];
#pragma unroll
  for (int r = 0; r < 8; ++r) { minval[r] = 3.0e38f; minidx[r] = 0; }

  const uint4* cb4 = (const uint4*)cb;   // 16 B = 8 bf16 granules

  for (int s = 0; s < N_STAGES; ++s) {
    const int cbase = s * CODES_PER_STAGE;

    // --- stage: swizzle 128 codes into B-fragment layout in LDS -----------
    __syncthreads();   // previous stage fully consumed
#pragma unroll
    for (int it = 0; it < 16; ++it) {
      const int j   = threadIdx.x + it * 256;   // uint4 index == frag layout
      const int l   = j & 31;
      const int ch  = (j >> 5) & 1;
      const int kc  = (j >> 6) & 7;
      const int ctl = (j >> 9) & 7;
      const int code = cbase + ctl * 16 + (l & 15);
      const int k    = kc * 32 + ch * 16 + (l >> 4) * 8;
      lds_b[j] = cb4[code * (E_DIM / 8) + (k >> 3)];
    }
    __syncthreads();

    // --- 8 column tiles of 16 codes from LDS ------------------------------
    for (int ctl = 0; ctl < 8; ++ctl) {
      const int code = cbase + ctl * 16 + lo;    // this lane's code column
      const float en = enorm[code];
      v8f acc = {};
#pragma unroll
      for (int kc = 0; kc < 8; ++kc) {
        const uint4* fp = lds_b + ((ctl * 8 + kc) * 2) * 32 + lane;
        uint4 c0 = fp[0];    // chunk 0: K = kc*32 + hi*8 + 0..7
        uint4 c1 = fp[32];   // chunk 1: K = kc*32 + 16 + hi*8 + 0..7
        v8bf b0 = __builtin_bit_cast(v8bf, c0);
        v8bf b1 = __builtin_bit_cast(v8bf, c1);
        v16bf B;
#pragma unroll
        for (int i = 0; i < 8; ++i) { B[i] = b0[i]; B[8 + i] = b1[i]; }
        acc = __builtin_amdgcn_wmma_f32_16x16x32_bf16(
            false, A[kc], false, B, (short)0, acc, false, false);
      }
      // D layout: VGPR r holds row M = r + 8*hi, column N = code.
      // argmin score: ||e||^2 - 2 z.e   (||z||^2 is row-constant).
#pragma unroll
      for (int r = 0; r < 8; ++r) {
        float sc = en - 2.0f * acc[r];
        if (sc < minval[r]) { minval[r] = sc; minidx[r] = code; }
      }
    }
  }

  // --- min-reduce across the 16 lanes of each half (tie -> lowest index)
#pragma unroll
  for (int r = 0; r < 8; ++r) {
#pragma unroll
    for (int m = 1; m <= 8; m <<= 1) {
      float ov = __shfl_xor(minval[r], m, 32);
      int   oi = __shfl_xor(minidx[r], m, 32);
      if (ov < minval[r] || (ov == minval[r] && oi < minidx[r])) {
        minval[r] = ov; minidx[r] = oi;
      }
    }
  }

  // --- write indices: lane hi*16+j (j<8) owns row rowBase + hi*8 + j
  if (lo < 8) {
    int sel = 0;
#pragma unroll
    for (int r = 0; r < 8; ++r) if (lo == r) sel = minidx[r];
    idx_out[rowBase + hi * 8 + lo] = (float)sel;
  }

  // --- gather z_q (exact fp32 codebook), write coalesced, accumulate SE
  float lsum = 0.0f;
#pragma unroll
  for (int j = 0; j < 8; ++j) {
    const int ia = __shfl(minidx[j], 0, 32);    // row j      (low half)
    const int ib = __shfl(minidx[j], 16, 32);   // row 8 + j  (high half)
#pragma unroll
    for (int h = 0; h < 2; ++h) {
      const int row = rowBase + h * 8 + j;
      const int ci  = h ? ib : ia;
      const float* er  = emb + (size_t)ci  * E_DIM + lane * 8;
      const float* zr2 = z   + (size_t)row * E_DIM + lane * 8;
      float*       qr  = zq  + (size_t)row * E_DIM + lane * 8;
#pragma unroll
      for (int i = 0; i < 8; ++i) {
        float e = er[i], x = zr2[i];
        qr[i] = e;
        float d = e - x;
        lsum += d * d;
      }
    }
  }

  // --- deterministic block partial (fixed-order, no fp atomics)
#pragma unroll
  for (int m = 1; m < 32; m <<= 1) lsum += __shfl_xor(lsum, m, 32);
  if (lane == 0) wsum[wave] = lsum;
  __syncthreads();
  if (threadIdx.x == 0) {
    float t = 0.0f;
#pragma unroll
    for (int w = 0; w < 8; ++w) t += wsum[w];
    partials[blockIdx.x] = t;
  }
}

// ---- final: fixed-tree reduce of 1024 partials -> losses ------------------
__global__ void vq_final_90787018703681(const float* __restrict__ partials,
                                        float* __restrict__ losses) {
  __shared__ float red[256];
  const int t = threadIdx.x;
  float s = 0.0f;
  for (int i = t; i < 1024; i += 256) s += partials[i];
  red[t] = s;
  __syncthreads();
  for (int k = 128; k > 0; k >>= 1) {
    if (t < k) red[t] += red[t + k];
    __syncthreads();
  }
  if (t == 0) {
    float mse = red[0] / (float)ZQ_ELEMS;
    losses[0] = mse;          // vq_loss
    losses[1] = BETA * mse;   // commitment_loss
  }
}

extern "C" void kernel_launch(void* const* d_in, const int* in_sizes, int n_in,
                              void* d_out, int out_size, void* d_ws, size_t ws_size,
                              hipStream_t stream) {
  (void)in_sizes; (void)n_in; (void)out_size; (void)ws_size;
  const float* z   = (const float*)d_in[0];   // [131072, 256]
  const float* emb = (const float*)d_in[1];   // [512, 256]

  char*   ws       = (char*)d_ws;
  __bf16* cb       = (__bf16*)ws;
  float*  enorm    = (float*)(ws + WS_ENORM_OFF);
  float*  partials = (float*)(ws + WS_PART_OFF);

  float* out    = (float*)d_out;
  float* zq     = out;                    // [0, 33554432)
  float* losses = out + ZQ_ELEMS;         // vq_loss, commitment_loss
  float* idxs   = out + ZQ_ELEMS + 2;     // [131072] indices (as f32)

  vq_prep_90787018703681<<<N_CODES, 256, 0, stream>>>(emb, cb, enorm);
  vq_main_90787018703681<<<N_ROWS / 128, 256, 0, stream>>>(
      z, emb, cb, enorm, zq, idxs, partials);
  vq_final_90787018703681<<<1, 256, 0, stream>>>(partials, losses);
}
